// NT_Xent_5454608466476
// MI455X (gfx1250) — matrix operand order repaired and checked by reading, not compiled, not run on hardware
//
#include <hip/hip_runtime.h>
#include <math.h>

typedef __attribute__((ext_vector_type(16))) __bf16 v16bf;
typedef __attribute__((ext_vector_type(8)))  __bf16 v8bf;
typedef __attribute__((ext_vector_type(4)))  __bf16 v4bf;
typedef __attribute__((ext_vector_type(8)))  float  v8f;
typedef __attribute__((ext_vector_type(4)))  float  v4f;
typedef __attribute__((ext_vector_type(4)))  unsigned int v4u;
typedef __attribute__((ext_vector_type(8)))  int    v8i;
typedef __attribute__((ext_vector_type(4)))  int    v4i;

#define BATCH   4096
#define NROW    8192
#define DIM     128
#define INV_T   10.0f
#define NEG_BIG (-1.0e30f)
#define NPART   8
#define COLS_PER_PART (NROW / NPART)   // 1024
#define NTILES  (COLS_PER_PART / 16)   // 64

#if defined(__HIP_DEVICE_COMPILE__) && __has_builtin(__builtin_amdgcn_tensor_load_to_lds)
#define USE_TDM 1
#else
#define USE_TDM 0
#endif

// ---------------------------------------------------------------------------
// Kernel 1: row-normalize z = [z_i; z_j] and split into bf16 hi/lo pair.
// ---------------------------------------------------------------------------
__global__ __launch_bounds__(256) void k_normalize(const float* __restrict__ zi,
                                                   const float* __restrict__ zj,
                                                   __bf16* __restrict__ zh,
                                                   __bf16* __restrict__ zl) {
    const int row  = blockIdx.x * (blockDim.x >> 5) + (threadIdx.x >> 5);
    const int lane = threadIdx.x & 31;
    if (row >= NROW) return;

    const float* src = (row < BATCH) ? (zi + (size_t)row * DIM)
                                     : (zj + (size_t)(row - BATCH) * DIM);
    v4f v = *(const v4f*)(src + lane * 4);
    float ss = v.x * v.x + v.y * v.y + v.z * v.z + v.w * v.w;
#pragma unroll
    for (int m = 16; m >= 1; m >>= 1) ss += __shfl_xor(ss, m, 32);

    const float scale = 1.0f / fmaxf(sqrtf(ss), 1e-8f);

    v4bf h, l;
#pragma unroll
    for (int e = 0; e < 4; ++e) {
        float f   = v[e] * scale;
        __bf16 hb = (__bf16)f;
        h[e] = hb;
        l[e] = (__bf16)(f - (float)hb);
    }
    *(v4bf*)(zh + (size_t)row * DIM + lane * 4) = h;
    *(v4bf*)(zl + (size_t)row * DIM + lane * 4) = l;
}

// ---------------------------------------------------------------------------
// TDM: 1-D D# descriptor moving one contiguous 4KB tile (16 rows x 256B)
// from global to LDS.  data_size=8B (code 3), tensor/tile dim0 = 512 units.
// Group0: count=1 | lds_addr | global_addr[56:0] | type=2.
// 6-arg builtin form (clang-23 / therock toolchain).
// ---------------------------------------------------------------------------
__device__ __forceinline__ void tdm_load_tile(const __bf16* gsrc, unsigned lds_off) {
#if USE_TDM
    uint64_t ga = (uint64_t)(uintptr_t)gsrc;
    v4u g0;
    g0.x = 1u;                                              // count=1 (valid)
    g0.y = lds_off;                                         // lds_addr
    g0.z = (unsigned)(ga & 0xffffffffu);                    // global_addr[31:0]
    g0.w = (unsigned)((ga >> 32) & 0x01ffffffu) | 0x80000000u; // addr[56:32]|type=2
    v8i g1;
    g1[0] = (3 << 16);                                      // data_size = 8B
    g1[1] = (int)(512u << 16);                              // tensor_dim0 = 512
    g1[2] = (int)(1u << 16);                                // tensor_dim1 = 1
    g1[3] = (int)(512u << 16);                              // tile_dim0 = 512
    g1[4] = 1;                                              // tile_dim1 = 1
    g1[5] = 512;                                            // tensor_dim0_stride
    g1[6] = 0;
    g1[7] = 0;
    v4i gz4 = {0, 0, 0, 0};
    v8i gz8 = {0, 0, 0, 0, 0, 0, 0, 0};
    __builtin_amdgcn_tensor_load_to_lds(g0, g1, gz4, gz4, gz8, 0);
#else
    (void)gsrc; (void)lds_off;
#endif
}

// ---------------------------------------------------------------------------
// Kernel 2: fused  S = (zn znT)/T  with online logsumexp over columns.
// Grid (NROW/128, NPART); block = 8 waves; wave = 16-row band.
// B tiles double-buffered in LDS, filled by TDM (or async-to-LDS fallback).
// ---------------------------------------------------------------------------
__global__ __launch_bounds__(256) void k_simloss_partial(
    const __bf16* __restrict__ zh, const __bf16* __restrict__ zl,
    float* __restrict__ pm, float* __restrict__ ps, float* __restrict__ pp) {

    __shared__ __align__(128) __bf16 sbh[2][16 * DIM];   // 8 KB
    __shared__ __align__(128) __bf16 sbl[2][16 * DIM];   // 8 KB

    const int lane    = threadIdx.x & 31;
    const int wid     = threadIdx.x >> 5;                 // 0..7
    const int rowBase = blockIdx.x * 128 + wid * 16;
    const int part    = blockIdx.y;                       // 0..NPART-1
    const int col0    = part * COLS_PER_PART;

    // ---- loop-invariant A fragments (this wave's 16 rows, K=0..127) ----
    const int ra = rowBase + (lane & 15);
    const int ak = (lane < 16) ? 0 : 16;
    v16bf ah[4], al[4];
#pragma unroll
    for (int s2 = 0; s2 < 4; ++s2) {
        const __bf16* ph = zh + (size_t)ra * DIM + s2 * 32 + ak;
        const __bf16* pl = zl + (size_t)ra * DIM + s2 * 32 + ak;
        v8bf h0 = *(const v8bf*)(ph);
        v8bf h1 = *(const v8bf*)(ph + 8);
        v8bf l0 = *(const v8bf*)(pl);
        v8bf l1 = *(const v8bf*)(pl + 8);
#pragma unroll
        for (int e = 0; e < 8; ++e) {
            ah[s2][e] = h0[e];  ah[s2][e + 8] = h1[e];
            al[s2][e] = l0[e];  al[s2][e + 8] = l1[e];
        }
    }

    float m[8], s[8], pos[8];
#pragma unroll
    for (int r = 0; r < 8; ++r) { m[r] = NEG_BIG; s[r] = 0.0f; pos[r] = NEG_BIG; }

    const int bn   = lane & 15;
    const int bk   = (lane < 16) ? 0 : 16;
    const int ioff = rowBase + ((lane >= 16) ? 8 : 0);

    // -------- staging helpers (TDM: wave 0 only; fallback: all threads) ----
#if USE_TDM
    #define STAGE(tile, buf)                                                   \
        if (wid == 0) {                                                        \
            const int _jc = col0 + (tile) * 16;                                \
            tdm_load_tile(zh + (size_t)_jc * DIM, (unsigned)(uintptr_t)&sbh[(buf)][0]); \
            tdm_load_tile(zl + (size_t)_jc * DIM, (unsigned)(uintptr_t)&sbl[(buf)][0]); \
        }
    #define WAIT_PENDING() if (wid == 0) __builtin_amdgcn_s_wait_tensorcnt(2)
    #define WAIT_ALL()     if (wid == 0) __builtin_amdgcn_s_wait_tensorcnt(0)
#else
    // thread t copies 16B chunk t of each matrix: 256 thr * 16B = 4KB
    #define STAGE(tile, buf)                                                   \
        {                                                                      \
            const int _jc    = col0 + (tile) * 16;                             \
            const int _row   = threadIdx.x >> 4;                               \
            const int _chunk = threadIdx.x & 15;                               \
            const int _off   = _row * DIM + _chunk * 8;                        \
            unsigned _dh = (unsigned)(uintptr_t)&sbh[(buf)][_off];             \
            unsigned _dl = (unsigned)(uintptr_t)&sbl[(buf)][_off];             \
            uint64_t _sh = (uint64_t)(uintptr_t)(zh + (size_t)(_jc + _row) * DIM + _chunk * 8); \
            uint64_t _sl = (uint64_t)(uintptr_t)(zl + (size_t)(_jc + _row) * DIM + _chunk * 8); \
            asm volatile("global_load_async_to_lds_b128 %0, %1, off"           \
                         :: "v"(_dh), "v"(_sh) : "memory");                    \
            asm volatile("global_load_async_to_lds_b128 %0, %1, off"           \
                         :: "v"(_dl), "v"(_sl) : "memory");                    \
        }
  #if defined(__HIP_DEVICE_COMPILE__)
    #define WAIT_PENDING() asm volatile("s_wait_asynccnt 0x2" ::: "memory")
    #define WAIT_ALL()     asm volatile("s_wait_asynccnt 0x0" ::: "memory")
  #else
    #define WAIT_PENDING()
    #define WAIT_ALL()
  #endif
#endif

#if defined(__HIP_DEVICE_COMPILE__) || USE_TDM
    STAGE(0, 0);
#endif

    for (int t = 0; t < NTILES; ++t) {
        const int buf = t & 1;
#if defined(__HIP_DEVICE_COMPILE__) || USE_TDM
        if (t + 1 < NTILES) {
            STAGE(t + 1, buf ^ 1);
            WAIT_PENDING();            // current tile's transfers complete
        } else {
            WAIT_ALL();
        }
#endif
        __syncthreads();               // tile t visible to all waves

        // ---- hoisted B fragments from LDS (ds_load clusters) ----
        v16bf bh[4], bl[4];
#pragma unroll
        for (int s2 = 0; s2 < 4; ++s2) {
            bh[s2] = *(const v16bf*)&sbh[buf][bn * DIM + s2 * 32 + bk];
            bl[s2] = *(const v16bf*)&sbl[buf][bn * DIM + s2 * 32 + bk];
        }

        v8f acc = {};
#pragma unroll
        for (int s2 = 0; s2 < 4; ++s2) {
            acc = __builtin_amdgcn_wmma_f32_16x16x32_bf16(false, ah[s2], false, bh[s2],
                                                          (short)0, acc, false, false);
            acc = __builtin_amdgcn_wmma_f32_16x16x32_bf16(false, al[s2], false, bh[s2],
                                                          (short)0, acc, false, false);
            acc = __builtin_amdgcn_wmma_f32_16x16x32_bf16(false, ah[s2], false, bl[s2],
                                                          (short)0, acc, false, false);
        }

        const int j = col0 + t * 16 + bn;
#pragma unroll
        for (int r = 0; r < 8; ++r) {
            const int i = ioff + r;
            float v = acc[r] * INV_T;
            if (j == i) v = NEG_BIG;                       // mask diagonal
            const int partner = (i < BATCH) ? (i + BATCH) : (i - BATCH);
            if (j == partner) pos[r] = v;                  // capture positive logit
            if (v <= m[r]) {
                s[r] += __expf(v - m[r]);
            } else {
                s[r] = s[r] * __expf(m[r] - v) + 1.0f;
                m[r] = v;
            }
        }
        __syncthreads();               // all reads of buf done before reuse
    }

    // ---- merge (m,s,pos) across the 16 lanes that share each row ----
#pragma unroll
    for (int mask = 1; mask <= 8; mask <<= 1) {
#pragma unroll
        for (int r = 0; r < 8; ++r) {
            float mo = __shfl_xor(m[r], mask, 32);
            float so = __shfl_xor(s[r], mask, 32);
            float po = __shfl_xor(pos[r], mask, 32);
            float mn = fmaxf(m[r], mo);
            s[r]   = s[r] * __expf(m[r] - mn) + so * __expf(mo - mn);
            m[r]   = mn;
            pos[r] = fmaxf(pos[r], po);
        }
    }

    if ((lane & 15) == 0) {                                // lanes 0 and 16 write
#pragma unroll
        for (int r = 0; r < 8; ++r) {
            const int i = ioff + r;
            const size_t idx = (size_t)i * NPART + part;
            pm[idx] = m[r];  ps[idx] = s[r];  pp[idx] = pos[r];
        }
    }
}

// ---------------------------------------------------------------------------
// Kernel 3: merge NPART partials per row -> per-row loss
// ---------------------------------------------------------------------------
__global__ __launch_bounds__(256) void k_combine(const float* __restrict__ pm,
                                                 const float* __restrict__ ps,
                                                 const float* __restrict__ pp,
                                                 float* __restrict__ loss) {
    const int i = blockIdx.x * blockDim.x + threadIdx.x;
    if (i >= NROW) return;
    float m = NEG_BIG, s = 0.0f, pos = NEG_BIG;
#pragma unroll
    for (int p = 0; p < NPART; ++p) {
        const size_t idx = (size_t)i * NPART + p;
        float mo = pm[idx], so = ps[idx];
        float mn = fmaxf(m, mo);
        s = s * __expf(m - mn) + so * __expf(mo - mn);
        m = mn;
        pos = fmaxf(pos, pp[idx]);
    }
    loss[i] = m + __logf(s) - pos;                         // CE_i, label 0
}

// ---------------------------------------------------------------------------
// Kernel 4: deterministic final sum / N
// ---------------------------------------------------------------------------
__global__ __launch_bounds__(256) void k_finalsum(const float* __restrict__ loss,
                                                  float* __restrict__ out) {
    __shared__ float red[256];
    float acc = 0.0f;
    for (int i = threadIdx.x; i < NROW; i += 256) acc += loss[i];
    red[threadIdx.x] = acc;
    __syncthreads();
#pragma unroll
    for (int st = 128; st > 0; st >>= 1) {
        if ((int)threadIdx.x < st) red[threadIdx.x] += red[threadIdx.x + st];
        __syncthreads();
    }
    if (threadIdx.x == 0) out[0] = red[0] / (float)NROW;
}

// ---------------------------------------------------------------------------
extern "C" void kernel_launch(void* const* d_in, const int* in_sizes, int n_in,
                              void* d_out, int out_size, void* d_ws, size_t ws_size,
                              hipStream_t stream) {
    (void)in_sizes; (void)n_in; (void)out_size; (void)ws_size;
    const float* zi = (const float*)d_in[0];
    const float* zj = (const float*)d_in[1];

    char* ws = (char*)d_ws;
    __bf16* zh = (__bf16*)ws;                                    // 2 MB
    __bf16* zl = (__bf16*)(ws + (size_t)NROW * DIM * 2);         // 2 MB
    float*  pm = (float*)(ws + (size_t)NROW * DIM * 4);          // 256 KB
    float*  psum = pm + (size_t)NROW * NPART;                    // 256 KB
    float*  pp = psum + (size_t)NROW * NPART;                    // 256 KB
    float*  loss = pp + (size_t)NROW * NPART;                    // 32 KB
    float*  out = (float*)d_out;

    k_normalize<<<NROW / 8, 256, 0, stream>>>(zi, zj, zh, zl);

    dim3 g2(NROW / 128, NPART);
    k_simloss_partial<<<g2, 256, 0, stream>>>(zh, zl, pm, psum, pp);

    k_combine<<<(NROW + 255) / 256, 256, 0, stream>>>(pm, psum, pp, loss);
    k_finalsum<<<1, 256, 0, stream>>>(loss, out);
}
